// CausalRpeAttn_88983132438700
// MI455X (gfx1250) — compile-verified, matrix operand order checked by reading, not run on hardware
//
#include <hip/hip_runtime.h>
#include <hip/hip_bf16.h>

#define D_MODEL   1024
#define NUM_HEADS 16
#define D_HEAD    64
#define SEQ       2048
#define BATCH     2
#define MTOT      (BATCH * SEQ)   // 4096 rows in the flattened GEMMs

typedef __bf16 bf16_t;
typedef __attribute__((ext_vector_type(16))) __bf16 v16bf;
typedef __attribute__((ext_vector_type(8)))  __bf16 v8bf;
typedef __attribute__((ext_vector_type(8)))  float   v8f;

__device__ __forceinline__ v8f wmma_bf16(v16bf a, v16bf b, v8f c) {
  // D = A(16x32 bf16) * B(32x16 bf16) + C(16x16 f32)
  return __builtin_amdgcn_wmma_f32_16x16x32_bf16(
      /*neg_a=*/false, a, /*neg_b=*/false, b,
      /*c_mod=*/(short)0, c, /*reuse_a=*/false, /*reuse_b=*/false);
}

// Loads a 16x32 bf16 fragment (A layout, or B layout with "row"=column of B)
// from a row-major matrix with leading dimension `ld` (elements).
// ISA 16-bit layout: lanes 0-15 rows 0-15 / K 0-7,16-23 ; lanes 16-31 K 8-15,24-31.
__device__ __forceinline__ v16bf load_frag16(const bf16_t* base, int ld, int lane) {
  const int row = lane & 15;
  const int k0  = (lane < 16) ? 0 : 8;
  const bf16_t* p = base + (size_t)row * ld + k0;
  v8bf lo = *(const v8bf*)(p);        // 16B contiguous
  v8bf hi = *(const v8bf*)(p + 16);   // 16B contiguous
  v16bf r;
#pragma unroll
  for (int i = 0; i < 8; ++i) { r[i] = lo[i]; r[i + 8] = hi[i]; }
  return r;
}

// ---- CDNA5 async copy global->LDS (ASYNCcnt-tracked), inline asm per ISA 15.18.3
__device__ __forceinline__ void async_load_b128(unsigned lds_off, unsigned goff,
                                                const void* sbase) {
  asm volatile("global_load_async_to_lds_b128 %0, %1, %2"
               :: "v"(lds_off), "v"(goff), "s"(sbase) : "memory");
}
__device__ __forceinline__ void wait_async0() {
  asm volatile("s_wait_asynccnt 0x0" ::: "memory");
}

// ---------------------------------------------------------------- converts
__global__ __launch_bounds__(256)
void cvt_f32_to_bf16(const float* __restrict__ in, bf16_t* __restrict__ out, int n) {
  int i = blockIdx.x * blockDim.x + threadIdx.x;
  if (i < n) out[i] = (bf16_t)in[i];
}

// ---------------------------------------------------------------- projections
// C[m,n] = sum_k A[m,k] * W[n,k] + bias[n]:
//   MODE 0: Q — RoPE + 1/sqrt(Dh), store bf16 [B,H,S,Dh]
//   MODE 1: K — RoPE,              store bf16 [B,H,S,Dh]
//   MODE 2: V —                    store bf16 [B,H,Dh,S] (transposed)
//   MODE 3: O —                    store f32  [M,N] (= d_out)
// One wave computes 16x64 (a full head wide), so rotate_half's dh±32 partner
// lives in a sibling accumulator of the SAME lane.
template <int MODE>
__global__ __launch_bounds__(128)
void proj_kernel(const bf16_t* __restrict__ A, const bf16_t* __restrict__ W,
                 const float* __restrict__ bias,
                 const float* __restrict__ cosb, const float* __restrict__ sinb,
                 bf16_t* __restrict__ outb, float* __restrict__ outf) {
  const int lane = threadIdx.x & 31;
  const int wave = threadIdx.x >> 5;
  const int n0 = blockIdx.x * 64;               // head-aligned column base
  const int m0 = blockIdx.y * 64 + wave * 16;   // row base

  v8f acc[4] = {v8f{}, v8f{}, v8f{}, v8f{}};
#pragma unroll 1
  for (int kk = 0; kk < D_MODEL; kk += 32) {
    if (kk + 64 < D_MODEL) {  // warm next chunk into cache (global_prefetch_b8)
      __builtin_prefetch(A + (size_t)m0 * D_MODEL + kk + 64, 0, 1);
      __builtin_prefetch(W + (size_t)n0 * D_MODEL + kk + 64, 0, 1);
    }
    v16bf a = load_frag16(A + (size_t)m0 * D_MODEL + kk, D_MODEL, lane);
#pragma unroll
    for (int t = 0; t < 4; ++t) {
      v16bf b = load_frag16(W + (size_t)(n0 + t * 16) * D_MODEL + kk, D_MODEL, lane);
      acc[t] = wmma_bf16(a, b, acc[t]);
    }
  }

  const int lane15 = lane & 15;
  const int half8  = (lane < 16) ? 0 : 8;
  const int h      = n0 >> 6;
  float bval[4];
#pragma unroll
  for (int t = 0; t < 4; ++t) bval[t] = bias[n0 + t * 16 + lane15];

#pragma unroll
  for (int v = 0; v < 8; ++v) {
    const int m    = m0 + v + half8;   // C layout: vgpr v -> row v / v+8
    const int bidx = m >> 11;
    const int s    = m & (SEQ - 1);
    float val[4];
#pragma unroll
    for (int t = 0; t < 4; ++t) val[t] = acc[t][v] + bval[t];

    if (MODE == 0 || MODE == 1) {
      const float scale = (MODE == 0) ? 0.125f : 1.0f;
#pragma unroll
      for (int t = 0; t < 4; ++t) {
        const int dh = t * 16 + lane15;
        const float c  = cosb[s * D_HEAD + dh];
        const float sn = sinb[s * D_HEAD + dh];
        const float rot = (t < 2) ? -val[t + 2] : val[t - 2];
        const float r = (val[t] * c + rot * sn) * scale;
        outb[(((size_t)bidx * NUM_HEADS + h) * SEQ + s) * D_HEAD + dh] = (bf16_t)r;
      }
    } else if (MODE == 2) {
#pragma unroll
      for (int t = 0; t < 4; ++t) {
        const int dh = t * 16 + lane15;
        outb[(((size_t)bidx * NUM_HEADS + h) * D_HEAD + dh) * SEQ + s] = (bf16_t)val[t];
      }
    } else {
#pragma unroll
      for (int t = 0; t < 4; ++t)
        outf[(size_t)m * D_MODEL + n0 + t * 16 + lane15] = val[t];
    }
  }
}

// ---------------------------------------------------------------- flash attention
// Block = 4 waves = 64 queries of one (b,h). K/V tiles (64 keys) are staged
// into LDS with async global->LDS copies, double-buffered, shared by all waves.
// LDS rows padded to 144B (36 dwords) -> bank-conflict-free 16B fragment reads.
#define KROWB 144            // padded row stride in bytes (72 bf16)
#define KROWE 72             // padded row stride in elements
#define KTILEB (64 * KROWB)  // 9216 bytes per 64x64 bf16 tile
#define SM_K0  0
#define SM_V0  (2 * KTILEB)            // 18432
#define SM_P0  (4 * KTILEB)            // 36864
#define SM_TOT (SM_P0 + 4 * 16 * KROWB)  // + 4 waves * 16x72 P tiles = 46080

__global__ __launch_bounds__(128)
void attn_kernel(const bf16_t* __restrict__ Qb, const bf16_t* __restrict__ Kb,
                 const bf16_t* __restrict__ Vtb, bf16_t* __restrict__ attnb) {
  __shared__ __align__(16) char smem[SM_TOT];

  const int tid  = threadIdx.x;
  const int lane = tid & 31;
  const int wave = tid >> 5;
  const int qg   = blockIdx.x % (SEQ / 64);
  const int bh   = blockIdx.x / (SEQ / 64);
  const int bidx = bh / NUM_HEADS;
  const int h    = bh % NUM_HEADS;
  const int q0   = qg * 64 + wave * 16;

  const bf16_t* Qh  = Qb  + (size_t)bh * SEQ * D_HEAD;   // [S,Dh]
  const bf16_t* Kh  = Kb  + (size_t)bh * SEQ * D_HEAD;   // [S,Dh]
  const bf16_t* Vth = Vtb + (size_t)bh * D_HEAD * SEQ;   // [Dh,S]

  const unsigned smb = (unsigned)(uintptr_t)(&smem[0]);  // LDS byte offset of smem

  // cooperative async stage of one 64-key K tile + V tile into buffer `buf`
  auto stage = [&](int buf, int kbase) {
    const unsigned kls = smb + SM_K0 + buf * KTILEB;
    const unsigned vls = smb + SM_V0 + buf * KTILEB;
#pragma unroll
    for (int i = 0; i < 4; ++i) {                 // 512 16B chunks / 128 threads
      const int idx = tid + i * 128;
      const int row = idx >> 3, chk = idx & 7;    // 64 rows x 8 chunks
      async_load_b128(kls + row * KROWB + chk * 16,
                      (unsigned)((kbase + row) * (D_HEAD * 2) + chk * 16), Kh);
    }
#pragma unroll
    for (int i = 0; i < 4; ++i) {
      const int idx = tid + i * 128;
      const int row = idx >> 3, chk = idx & 7;    // row = dh, chunks over keys
      async_load_b128(vls + row * KROWB + chk * 16,
                      (unsigned)(row * (SEQ * 2) + kbase * 2 + chk * 16), Vth);
    }
  };

  // Q fragments: rows = queries, K-dim = dh (two 32-wide chunks of Dh=64)
  const v16bf aq0 = load_frag16(Qh + (size_t)q0 * D_HEAD,      D_HEAD, lane);
  const v16bf aq1 = load_frag16(Qh + (size_t)q0 * D_HEAD + 32, D_HEAD, lane);

  const int lane15 = lane & 15;
  const int half8  = (lane < 16) ? 0 : 8;

  float mrun[8], lrun[8];
  v8f acc[4] = {v8f{}, v8f{}, v8f{}, v8f{}};
#pragma unroll
  for (int v = 0; v < 8; ++v) { mrun[v] = -3.0e38f; lrun[v] = 0.0f; }

  const int nkb = qg + 1;          // 64-key blocks, causal bound (uniform in block)
  stage(0, 0);

  for (int kb = 0; kb < nkb; ++kb) {
    const int cur = kb & 1;
    const int kbase = kb * 64;
    wait_async0();                 // my stage of block kb is complete
    __syncthreads();               // everyone's stage visible; prev buffer free
    if (kb + 1 < nkb) stage(cur ^ 1, (kb + 1) * 64);

    const bf16_t* Kls = (const bf16_t*)&smem[SM_K0 + cur * KTILEB];
    const bf16_t* Vls = (const bf16_t*)&smem[SM_V0 + cur * KTILEB];
    bf16_t*       Pls = (bf16_t*)&smem[SM_P0 + wave * 16 * KROWB];

    // ---- scores: 4 subtiles of 16 keys; diagonal block prunes t > wave
    v8f sarr[4];
    bool valid[4];
#pragma unroll
    for (int t = 0; t < 4; ++t) {
      valid[t] = (kbase + t * 16) <= (q0 + 15);
      v8f s = {};
      if (valid[t]) {
        v16bf b0 = load_frag16(Kls + (t * 16) * KROWE,      KROWE, lane);
        v16bf b1 = load_frag16(Kls + (t * 16) * KROWE + 32, KROWE, lane);
        s = wmma_bf16(aq0, b0, s);
        s = wmma_bf16(aq1, b1, s);
      }
      sarr[t] = s;
    }

    // ---- causal mask + row max (row lives across one 16-lane half)
    float rstat[8];
#pragma unroll
    for (int v = 0; v < 8; ++v) {
      const int rowg = q0 + v + half8;
      rstat[v] = -3.0e38f;
#pragma unroll
      for (int t = 0; t < 4; ++t) {
        if (!valid[t] || kbase + t * 16 + lane15 > rowg) sarr[t][v] = -3.0e38f;
        rstat[v] = fmaxf(rstat[v], sarr[t][v]);
      }
    }
#pragma unroll
    for (int v = 0; v < 8; ++v)
#pragma unroll
      for (int msk = 1; msk <= 8; msk <<= 1)
        rstat[v] = fmaxf(rstat[v], __shfl_xor(rstat[v], msk, 32));

    float mnew[8];
#pragma unroll
    for (int v = 0; v < 8; ++v) {
      mnew[v] = fmaxf(mrun[v], rstat[v]);          // finite: diag col present
      const float alpha = __expf(mrun[v] - mnew[v]);
      mrun[v] = mnew[v];
      lrun[v] *= alpha;
      rstat[v] = 0.0f;                             // becomes row-sum
#pragma unroll
      for (int d = 0; d < 4; ++d) acc[d][v] *= alpha;
    }

    // ---- P = exp(S - m): accumulate row-sum, store bf16 tile for A-frag reload
#pragma unroll
    for (int t = 0; t < 4; ++t)
#pragma unroll
      for (int v = 0; v < 8; ++v) {
        const float p = __expf(sarr[t][v] - mnew[v]);   // masked -> 0
        rstat[v] += p;
        Pls[(v + half8) * KROWE + t * 16 + lane15] = (bf16_t)p;
      }
#pragma unroll
    for (int v = 0; v < 8; ++v) {
#pragma unroll
      for (int msk = 1; msk <= 8; msk <<= 1)
        rstat[v] += __shfl_xor(rstat[v], msk, 32);
      lrun[v] += rstat[v];
    }

    asm volatile("s_wait_dscnt 0x0" ::: "memory");   // per-wave P relayout ready
    v16bf ap0 = load_frag16(Pls,      KROWE, lane);  // keys 0..31 of block
    v16bf ap1 = load_frag16(Pls + 32, KROWE, lane);  // keys 32..63

    // ---- acc(16xDh) += P(16x64) * V(64xDh)
#pragma unroll
    for (int d = 0; d < 4; ++d) {
      v16bf bv0 = load_frag16(Vls + (d * 16) * KROWE,      KROWE, lane);
      v16bf bv1 = load_frag16(Vls + (d * 16) * KROWE + 32, KROWE, lane);
      acc[d] = wmma_bf16(ap0, bv0, acc[d]);
      acc[d] = wmma_bf16(ap1, bv1, acc[d]);
    }
  }

  // ---- normalize, store bf16 [B,S,D] (heads concatenated) for the O-proj
#pragma unroll
  for (int v = 0; v < 8; ++v) {
    const float inv = 1.0f / lrun[v];
    const int q = q0 + v + half8;
#pragma unroll
    for (int d = 0; d < 4; ++d) {
      const float o = acc[d][v] * inv;
      attnb[((size_t)bidx * SEQ + q) * D_MODEL + h * D_HEAD + d * 16 + lane15] =
          (bf16_t)o;
    }
  }
}

// ---------------------------------------------------------------- launch
extern "C" void kernel_launch(void* const* d_in, const int* in_sizes, int n_in,
                              void* d_out, int out_size, void* d_ws, size_t ws_size,
                              hipStream_t stream) {
  (void)in_sizes; (void)n_in; (void)out_size; (void)ws_size;
  const float* qkv  = (const float*)d_in[0];
  const float* cosb = (const float*)d_in[1];
  const float* sinb = (const float*)d_in[2];
  const float* Wq   = (const float*)d_in[3];
  const float* bq   = (const float*)d_in[4];
  const float* Wk   = (const float*)d_in[5];
  const float* bk   = (const float*)d_in[6];
  const float* Wv   = (const float*)d_in[7];
  const float* bv   = (const float*)d_in[8];
  const float* Wo   = (const float*)d_in[9];
  const float* bo   = (const float*)d_in[10];
  float* out = (float*)d_out;

  char* ws = (char*)d_ws;
  size_t off = 0;
  auto alloc = [&](size_t bytes) -> char* {
    char* p = ws + off;
    off += (bytes + 255) & ~(size_t)255;
    return p;
  };
  const size_t xN = (size_t)MTOT * D_MODEL;
  const size_t wN = (size_t)D_MODEL * D_MODEL;
  bf16_t* qkv_bf = (bf16_t*)alloc(xN * 2);
  bf16_t* wq_bf  = (bf16_t*)alloc(wN * 2);
  bf16_t* wk_bf  = (bf16_t*)alloc(wN * 2);
  bf16_t* wv_bf  = (bf16_t*)alloc(wN * 2);
  bf16_t* wo_bf  = (bf16_t*)alloc(wN * 2);
  bf16_t* Qbuf   = (bf16_t*)alloc(xN * 2);   // [B,H,S,Dh]
  bf16_t* Kbuf   = (bf16_t*)alloc(xN * 2);   // [B,H,S,Dh]
  bf16_t* Vtbuf  = (bf16_t*)alloc(xN * 2);   // [B,H,Dh,S]
  bf16_t* attnb  = (bf16_t*)alloc(xN * 2);   // [B,S,D]

  cvt_f32_to_bf16<<<(int)((xN + 255) / 256), 256, 0, stream>>>(qkv, qkv_bf, (int)xN);
  cvt_f32_to_bf16<<<(int)((wN + 255) / 256), 256, 0, stream>>>(Wq, wq_bf, (int)wN);
  cvt_f32_to_bf16<<<(int)((wN + 255) / 256), 256, 0, stream>>>(Wk, wk_bf, (int)wN);
  cvt_f32_to_bf16<<<(int)((wN + 255) / 256), 256, 0, stream>>>(Wv, wv_bf, (int)wN);
  cvt_f32_to_bf16<<<(int)((wN + 255) / 256), 256, 0, stream>>>(Wo, wo_bf, (int)wN);

  dim3 pgrid(D_MODEL / 64, MTOT / 64);
  proj_kernel<0><<<pgrid, 128, 0, stream>>>(qkv_bf, wq_bf, bq, cosb, sinb, Qbuf,  nullptr);
  proj_kernel<1><<<pgrid, 128, 0, stream>>>(qkv_bf, wk_bf, bk, cosb, sinb, Kbuf,  nullptr);
  proj_kernel<2><<<pgrid, 128, 0, stream>>>(qkv_bf, wv_bf, bv, cosb, sinb, Vtbuf, nullptr);

  attn_kernel<<<BATCH * NUM_HEADS * (SEQ / 64), 128, 0, stream>>>(Qbuf, Kbuf, Vtbuf, attnb);

  proj_kernel<3><<<pgrid, 128, 0, stream>>>(attnb, wo_bf, bo, nullptr, nullptr, nullptr, out);
}
